// CustomMultiLossLayer_29308856828132
// MI455X (gfx1250) — compile-verified, matrix operand order checked by reading, not run on hardware
//
#include <hip/hip_runtime.h>
#include <hip/hip_bf16.h>

typedef __attribute__((ext_vector_type(4)))  float    v4f;
typedef __attribute__((ext_vector_type(8)))  float    v8f;
typedef __attribute__((ext_vector_type(16))) _Float16 v16h;

#define T_STEPS 500
#define N_ROWS  32768
#define NITEMS  ((size_t)T_STEPS * (size_t)N_ROWS)   // 16,384,000 items per task
#define NGROUPS (NITEMS / 4)                         // 4,096,000 groups (4 items = 12 floats each)

// ---------------------------------------------------------------------------
// Wave32 sum-reduction via V_WMMA_F32_16X16X32_F16 with a ones B-matrix.
// Each lane places its fp32 partial (as f16) into one unique A element; with
// B = ones, D[m][n] = rowsum(A). Lane L holds D[v + 8*(L>=16)][L%16] in VGPR v,
// so summing the 8 D registers gives half-row-range sums; one shfl_xor(16)
// completes the full 32-lane sum (accumulation itself happens in fp32 inside
// the matrix core).  EXEC must be all ones: callers invoke this convergently.
// ---------------------------------------------------------------------------
__device__ __forceinline__ float wave_sum_wmma(float x) {
  v16h a = {};
  a[0] = (_Float16)x;
  v16h ones;
#pragma unroll
  for (int i = 0; i < 16; ++i) ones[i] = (_Float16)1.0f;
  v8f c = {};
  // 8 args: (neg_a, A, neg_b, B, c_mod, C, reuse_a, reuse_b)
  v8f d = __builtin_amdgcn_wmma_f32_16x16x32_f16(false, a, false, ones,
                                                 (short)0, c, false, false);
  float s = 0.0f;
#pragma unroll
  for (int i = 0; i < 8; ++i) s += d[i];
  s += __shfl_xor(s, 16, 32);
  return s;
}

__global__ void zero_accum_kernel(float* __restrict__ accum) {
  if (threadIdx.x < 2) accum[threadIdx.x] = 0.0f;
}

// ---------------------------------------------------------------------------
// Streaming MC cross-entropy partial-sum kernel.
// Blocks [0, grid/2) -> task 0, [grid/2, grid) -> task 1.
// Each thread: grid-stride over groups of 4 (t,n) items = 12 contiguous eps
// floats = 3x b128 nontemporal loads (HBM stream, 393 MB total).
// y_pred/y_true re-reads (28 B/item) are L2-resident (2 MB working set).
// ---------------------------------------------------------------------------
__global__ __launch_bounds__(256) void hetero_mc_kernel(
    const float* __restrict__ yt0, const float* __restrict__ yp0,
    const float* __restrict__ yt1, const float* __restrict__ yp1,
    const float* __restrict__ eps0, const float* __restrict__ eps1,
    float* __restrict__ accum) {
  const unsigned half = gridDim.x >> 1;
  const int task = (blockIdx.x >= half) ? 1 : 0;
  const unsigned blk = blockIdx.x - (task ? half : 0u);

  const float* __restrict__ yt = task ? yt1 : yt0;
  const v4f*   __restrict__ yp = (const v4f*)(task ? yp1 : yp0);
  const float* __restrict__ ep = task ? eps1 : eps0;

  const size_t stride = (size_t)half * (size_t)blockDim.x;
  float acc = 0.0f;

  for (size_t g = (size_t)blk * blockDim.x + threadIdx.x; g < NGROUPS; g += stride) {
    const v4f* p = (const v4f*)(ep + g * 12);   // 48B-aligned
    // Speculative L2 prefetch one grid-stride ahead (~12.6 MB/round << 192 MB L2).
    __builtin_prefetch((const void*)(ep + (g + stride) * 12), 0, 0);

    v4f e0 = __builtin_nontemporal_load(p + 0);
    v4f e1 = __builtin_nontemporal_load(p + 1);
    v4f e2 = __builtin_nontemporal_load(p + 2);

    float f[12];
#pragma unroll
    for (int i = 0; i < 4; ++i) { f[i] = e0[i]; f[4 + i] = e1[i]; f[8 + i] = e2[i]; }

    const int n0 = (int)((g * 4) & (size_t)(N_ROWS - 1));  // 4 | N_ROWS: no t-boundary straddle
#pragma unroll
    for (int i = 0; i < 4; ++i) {
      const int n = n0 + i;
      const v4f pr = yp[n];                     // {l0,l1,l2, logvar-head}
      const float t0 = yt[3 * n + 0];
      const float t1 = yt[3 * n + 1];
      const float t2 = yt[3 * n + 2];
      const float s  = __expf(0.5f * pr[3]);    // sqrt(exp(x)) = exp(x/2)
      const float d0 = fmaf(f[3 * i + 0], s, pr[0]);
      const float d1 = fmaf(f[3 * i + 1], s, pr[1]);
      const float d2 = fmaf(f[3 * i + 2], s, pr[2]);
      const float m  = fmaxf(d0, fmaxf(d1, d2));
      const float lse = m + __logf(__expf(d0 - m) + __expf(d1 - m) + __expf(d2 - m));
      // ce = sum_i t_i*(lse - d_i)
      acc += (t0 + t1 + t2) * lse - (t0 * d0 + t1 * d1 + t2 * d2);
    }
  }

  // Convergent reduction: wave via WMMA, block via LDS, grid via global atomic.
  const float wsum = wave_sum_wmma(acc);
  __shared__ float bsum;
  if (threadIdx.x == 0) bsum = 0.0f;
  __syncthreads();
  if ((threadIdx.x & 31) == 0) atomicAdd(&bsum, wsum);
  __syncthreads();
  if (threadIdx.x == 0) atomicAdd(&accum[task], bsum);
}

__global__ void finalize_kernel(const float* __restrict__ accum,
                                const float* __restrict__ log_vars,
                                float* __restrict__ out) {
  const float inv = 1.0f / (float)NITEMS;
  const float lv0 = log_vars[0];
  const float lv1 = log_vars[1];
  out[0] = __expf(-lv0) * (accum[0] * inv) + lv0 +
           __expf(-lv1) * (accum[1] * inv) + lv1;
}

extern "C" void kernel_launch(void* const* d_in, const int* in_sizes, int n_in,
                              void* d_out, int out_size, void* d_ws, size_t ws_size,
                              hipStream_t stream) {
  (void)in_sizes; (void)n_in; (void)out_size; (void)ws_size;
  const float* yt0 = (const float*)d_in[0];
  const float* yp0 = (const float*)d_in[1];
  const float* yt1 = (const float*)d_in[2];
  const float* yp1 = (const float*)d_in[3];
  const float* lv  = (const float*)d_in[4];
  const float* e0  = (const float*)d_in[5];
  const float* e1  = (const float*)d_in[6];
  float* accum = (float*)d_ws;   // 2 fp32 task accumulators (workspace is poisoned: re-zero each call)
  float* out   = (float*)d_out;

  zero_accum_kernel<<<1, 32, 0, stream>>>(accum);
  // 2048 blocks x 256 threads (8 wave32/block): 1024 blocks per task,
  // 262,144 threads/task over 4,096,000 groups (~15.6 groups/thread).
  hetero_mc_kernel<<<2048, 256, 0, stream>>>(yt0, yp0, yt1, yp1, e0, e1, accum);
  finalize_kernel<<<1, 1, 0, stream>>>(accum, lv, out);
}
